// BPTTSigmoidDiagonalRNN_72576357368245
// MI455X (gfx1250) — matrix operand order repaired and checked by reading, not compiled, not run on hardware
//
#include <hip/hip_runtime.h>

typedef __attribute__((ext_vector_type(16))) __bf16 v16bf;
typedef __attribute__((ext_vector_type(8)))  __bf16 v8bf;
typedef __attribute__((ext_vector_type(4)))  __bf16 v4bf;
typedef __attribute__((ext_vector_type(8)))  float  v8f;
typedef __attribute__((ext_vector_type(4)))  float  v4f;

constexpr int T = 8192;
constexpr int H = 2048;
constexpr int D = 1024;
constexpr int NC = 256;        // scan chunks
constexpr int CHUNK = T / NC;  // 32 steps per chunk
constexpr int H4 = H / 4;      // float4 channel groups

// ---------- bf16 split helpers (hi = truncation -> exact residual) ----------
__device__ __forceinline__ __bf16 bf_trunc(float x, float& rem) {
    unsigned u  = __builtin_bit_cast(unsigned, x);
    unsigned hi = u & 0xFFFF0000u;
    rem = x - __builtin_bit_cast(float, hi);
    return __builtin_bit_cast(__bf16, (unsigned short)(hi >> 16));
}
__device__ __forceinline__ __bf16 bf_rne(float x) {
    unsigned u = __builtin_bit_cast(unsigned, x);
    u += 0x7FFFu + ((u >> 16) & 1u);
    return __builtin_bit_cast(__bf16, (unsigned short)(u >> 16));
}

__device__ __forceinline__ float sigm(float x) {
    return 1.0f / (1.0f + __expf(-x));
}
__device__ __forceinline__ v4f sigm4(v4f x) {
    v4f r;
    r.x = sigm(x.x); r.y = sigm(x.y); r.z = sigm(x.z); r.w = sigm(x.w);
    return r;
}

__device__ __forceinline__ v16bf cat16(v8bf a, v8bf b) {
    return __builtin_shufflevector(a, b, 0, 1, 2, 3, 4, 5, 6, 7,
                                         8, 9, 10, 11, 12, 13, 14, 15);
}

// ============================================================================
// Pre-pass: split fp32 rows (length D) into row-interleaved bf16 hi/lo:
//   dst[(row*2 + 0)*D + col] = hi,  dst[(row*2 + 1)*D + col] = lo
// lo row is a fixed +2048B from hi row -> single base + immediate in the GEMM.
// ============================================================================
__global__ __launch_bounds__(256)
void split_bf16_rows_kernel(const float* __restrict__ src,
                            __bf16* __restrict__ dst, int n4) {
    int i = blockIdx.x * blockDim.x + threadIdx.x;
    if (i >= n4) return;
    const int col4 = i & (D / 4 - 1);            // 4-elem group within row
    const int row  = i >> 8;                     // i / (D/4), D/4 == 256
    v4f x = ((const v4f*)src)[i];
    v4bf h, l;
#pragma unroll
    for (int e = 0; e < 4; ++e) {
        float rem;
        h[e] = bf_trunc(x[e], rem);
        l[e] = bf_rne(rem);
    }
    ((v4bf*)(dst + (size_t)(row * 2 + 0) * D))[col4] = h;
    ((v4bf*)(dst + (size_t)(row * 2 + 1) * D))[col4] = l;
}

// ============================================================================
// GEMM from row-interleaved split planes. 6 loop-carried base pointers,
// K unrolled x4 -> all fragment loads use immediate offsets.
// Block = 256 thr (8 waves: 2 along M x 4 along N) -> 128x128 block tile.
// Wave tile = 64x32 (4 Mtiles x 2 Ntiles); 24 WMMAs per K-step.
// ============================================================================
__global__ __launch_bounds__(256)
void rnn_gemm_split_kernel(const __bf16* __restrict__ Xs,
                           const __bf16* __restrict__ Bs,
                           float* __restrict__ U) {
    const int lane   = threadIdx.x & 31;
    const int wave   = threadIdx.x >> 5;
    const int wm     = wave >> 2;                 // 0..1
    const int wn     = wave & 3;                  // 0..3
    const int m0     = blockIdx.x * 128 + wm * 64;
    const int n0     = blockIdx.y * 128 + wn * 32;
    const int r      = lane & 15;
    const int laneHi = lane >> 4;
    const int kgA    = laneHi * 8;    // A: 8-elem runs at +0 and +16
    const int kgB    = laneHi * 8;    // B: 8-elem runs at +0 and +16

    const __bf16* ap0[4];
#pragma unroll
    for (int mt = 0; mt < 4; ++mt)
        ap0[mt] = Xs + (size_t)(m0 + mt * 16 + r) * 2 * D + kgA;
    const __bf16* bp0[2];
#pragma unroll
    for (int nt = 0; nt < 2; ++nt)
        bp0[nt] = Bs + (size_t)(n0 + nt * 16 + r) * 2 * D + kgB;

    v8f acc[4][2];
#pragma unroll
    for (int mt = 0; mt < 4; ++mt)
#pragma unroll
        for (int nt = 0; nt < 2; ++nt)
#pragma unroll
            for (int e = 0; e < 8; ++e) acc[mt][nt][e] = 0.0f;

    for (int kb = 0; kb < D; kb += 128) {
#pragma unroll
        for (int ku = 0; ku < 4; ++ku) {
            const int k0 = ku * 32;               // compile-time immediate
            v16bf bh[2], bl[2];
#pragma unroll
            for (int nt = 0; nt < 2; ++nt) {
                const __bf16* p = bp0[nt] + kb + k0;
                bh[nt] = cat16(*(const v8bf*)(p),     *(const v8bf*)(p + 16));
                bl[nt] = cat16(*(const v8bf*)(p + D), *(const v8bf*)(p + D + 16));
            }
#pragma unroll
            for (int mt = 0; mt < 4; ++mt) {
                const __bf16* p = ap0[mt] + kb + k0;
                v16bf ah = cat16(*(const v8bf*)(p),     *(const v8bf*)(p + 16));
                v16bf al = cat16(*(const v8bf*)(p + D), *(const v8bf*)(p + D + 16));
#pragma unroll
                for (int nt = 0; nt < 2; ++nt) {
                    acc[mt][nt] = __builtin_amdgcn_wmma_f32_16x16x32_bf16(
                        false, ah, false, bh[nt], (short)0, acc[mt][nt], false, false);
                    acc[mt][nt] = __builtin_amdgcn_wmma_f32_16x16x32_bf16(
                        false, ah, false, bl[nt], (short)0, acc[mt][nt], false, false);
                    acc[mt][nt] = __builtin_amdgcn_wmma_f32_16x16x32_bf16(
                        false, al, false, bh[nt], (short)0, acc[mt][nt], false, false);
                }
            }
        }
    }

    // ---- store C/D: element e -> row e + 8*laneHi, col r ----
#pragma unroll
    for (int mt = 0; mt < 4; ++mt)
#pragma unroll
        for (int nt = 0; nt < 2; ++nt) {
            float* op = U + (size_t)(m0 + mt * 16 + 8 * laneHi) * H
                          + (n0 + nt * 16 + r);
#pragma unroll
            for (int e = 0; e < 8; ++e) op[(size_t)e * H] = acc[mt][nt][e];
        }
}

// ============================================================================
// Fallback GEMM (fused conversion) when d_ws is too small for split planes.
// ============================================================================
__global__ __launch_bounds__(256)
void rnn_gemm_fused_kernel(const float* __restrict__ X, const float* __restrict__ Bm,
                           float* __restrict__ U) {
    const int lane   = threadIdx.x & 31;
    const int wave   = threadIdx.x >> 5;
    const int wm     = wave >> 2;
    const int wn     = wave & 3;
    const int m0     = blockIdx.x * 128 + wm * 64;
    const int n0     = blockIdx.y * 128 + wn * 32;
    const int r      = lane & 15;
    const int laneHi = lane >> 4;
    const int kgA    = laneHi * 8;
    const int kgB    = laneHi * 8;

    v8f acc[4][2];
#pragma unroll
    for (int mt = 0; mt < 4; ++mt)
#pragma unroll
        for (int nt = 0; nt < 2; ++nt)
#pragma unroll
            for (int e = 0; e < 8; ++e) acc[mt][nt][e] = 0.0f;

    for (int k0 = 0; k0 < D; k0 += 32) {
        v16bf bh[2], bl[2];
#pragma unroll
        for (int nt = 0; nt < 2; ++nt) {
            const float* bp = Bm + (size_t)(n0 + nt * 16 + r) * D + k0 + kgB;
#pragma unroll
            for (int qd = 0; qd < 4; ++qd) {
                v4f q = *(const v4f*)(bp + (qd >> 1) * 16 + (qd & 1) * 4);
#pragma unroll
                for (int e = 0; e < 4; ++e) {
                    float rem;
                    bh[nt][qd * 4 + e] = bf_trunc(q[e], rem);
                    bl[nt][qd * 4 + e] = bf_rne(rem);
                }
            }
        }
#pragma unroll
        for (int mt = 0; mt < 4; ++mt) {
            const float* ap = X + (size_t)(m0 + mt * 16 + r) * D + k0 + kgA;
            v16bf ah, al;
#pragma unroll
            for (int g = 0; g < 2; ++g) {
                v4f p0 = *(const v4f*)(ap + g * 16);
                v4f p1 = *(const v4f*)(ap + g * 16 + 4);
#pragma unroll
                for (int e = 0; e < 4; ++e) {
                    float rem;
                    ah[g * 8 + e] = bf_trunc(p0[e], rem);
                    al[g * 8 + e] = bf_rne(rem);
                    float rem2;
                    ah[g * 8 + 4 + e] = bf_trunc(p1[e], rem2);
                    al[g * 8 + 4 + e] = bf_rne(rem2);
                }
            }
#pragma unroll
            for (int nt = 0; nt < 2; ++nt) {
                acc[mt][nt] = __builtin_amdgcn_wmma_f32_16x16x32_bf16(
                    false, ah, false, bh[nt], (short)0, acc[mt][nt], false, false);
                acc[mt][nt] = __builtin_amdgcn_wmma_f32_16x16x32_bf16(
                    false, ah, false, bl[nt], (short)0, acc[mt][nt], false, false);
                acc[mt][nt] = __builtin_amdgcn_wmma_f32_16x16x32_bf16(
                    false, al, false, bh[nt], (short)0, acc[mt][nt], false, false);
            }
        }
    }
#pragma unroll
    for (int mt = 0; mt < 4; ++mt)
#pragma unroll
        for (int nt = 0; nt < 2; ++nt) {
            float* op = U + (size_t)(m0 + mt * 16 + 8 * laneHi) * H
                          + (n0 + nt * 16 + r);
#pragma unroll
            for (int e = 0; e < 8; ++e) op[(size_t)e * H] = acc[mt][nt][e];
        }
}

// ============================================================================
// Scan phase 1: per-chunk local state with zero init -> carry buffer S[c,h]
// ============================================================================
__global__ __launch_bounds__(256)
void scan_local_kernel(const float* __restrict__ U, const float* __restrict__ lam,
                       float* __restrict__ Scar) {
    const int h4 = blockIdx.y * blockDim.x + threadIdx.x;   // 0..H4-1
    const int c  = blockIdx.x;
    const v4f a  = sigm4(((const v4f*)lam)[h4]);
    const v4f* up = (const v4f*)U + (size_t)c * CHUNK * H4 + h4;
    v4f s = {0.0f, 0.0f, 0.0f, 0.0f};
#pragma unroll 4
    for (int i = 0; i < CHUNK; ++i) {
        v4f u = up[(size_t)i * H4];
        s = a * s + u;
    }
    ((v4f*)Scar)[(size_t)c * H4 + h4] = s;
}

// ============================================================================
// Scan phase 2: exclusive carry scan across chunks with a^CHUNK (in place)
// ============================================================================
__global__ __launch_bounds__(256)
void scan_carry_kernel(const float* __restrict__ lam, float* __restrict__ Scar) {
    const int h4 = blockIdx.x * blockDim.x + threadIdx.x;   // 0..H4-1
    const v4f a  = sigm4(((const v4f*)lam)[h4]);
    v4f aL = a;
#pragma unroll
    for (int s = 0; s < 5; ++s) aL = aL * aL;               // a^32 == a^CHUNK
    v4f* sp = (v4f*)Scar + h4;
    v4f g = {0.0f, 0.0f, 0.0f, 0.0f};
    for (int c = 0; c < NC; ++c) {
        v4f s = sp[(size_t)c * H4];
        sp[(size_t)c * H4] = g;                              // exclusive carry
        g = aL * g + s;
    }
}

// ============================================================================
// Scan phase 3: re-run each chunk seeded with its carry, overwrite U in place
// ============================================================================
__global__ __launch_bounds__(256)
void scan_apply_kernel(float* __restrict__ U, const float* __restrict__ lam,
                       const float* __restrict__ Scar) {
    const int h4 = blockIdx.y * blockDim.x + threadIdx.x;
    const int c  = blockIdx.x;
    const v4f a  = sigm4(((const v4f*)lam)[h4]);
    v4f g = ((const v4f*)Scar)[(size_t)c * H4 + h4];
    v4f* up = (v4f*)U + (size_t)c * CHUNK * H4 + h4;
#pragma unroll 4
    for (int i = 0; i < CHUNK; ++i) {
        v4f u = up[(size_t)i * H4];
        g = a * g + u;
        up[(size_t)i * H4] = g;
    }
}

extern "C" void kernel_launch(void* const* d_in, const int* in_sizes, int n_in,
                              void* d_out, int out_size, void* d_ws, size_t ws_size,
                              hipStream_t stream) {
    const float* X   = (const float*)d_in[0];   // [T, D]
    const float* lam = (const float*)d_in[1];   // [H]
    const float* Bm  = (const float*)d_in[2];   // [H, D]
    float* out = (float*)d_out;                 // [T, H]

    const size_t xElems = (size_t)T * D;
    const size_t bElems = (size_t)H * D;
    const size_t needed = 2 * xElems * sizeof(__bf16)
                        + 2 * bElems * sizeof(__bf16)
                        + (size_t)NC * H * sizeof(float);

    float* Scar;
    dim3 gGemm(T / 128, H / 128);               // 64 x 16

    if (ws_size >= needed) {
        __bf16* Xs = (__bf16*)d_ws;             // [T][2][D] hi/lo interleaved
        __bf16* Bs = Xs + 2 * xElems;           // [H][2][D] hi/lo interleaved
        Scar = (float*)(Bs + 2 * bElems);

        const int xN4 = (int)(xElems / 4);
        const int bN4 = (int)(bElems / 4);
        split_bf16_rows_kernel<<<(xN4 + 255) / 256, 256, 0, stream>>>(X, Xs, xN4);
        split_bf16_rows_kernel<<<(bN4 + 255) / 256, 256, 0, stream>>>(Bm, Bs, bN4);
        rnn_gemm_split_kernel<<<gGemm, 256, 0, stream>>>(Xs, Bs, out);
    } else {
        Scar = (float*)d_ws;                    // needs only 2 MiB
        rnn_gemm_fused_kernel<<<gGemm, 256, 0, stream>>>(X, Bm, out);
    }

    dim3 gScan(NC, H4 / 256);                   // 256 x 2
    scan_local_kernel<<<gScan, 256, 0, stream>>>(out, lam, Scar);
    scan_carry_kernel<<<dim3(H4 / 256), 256, 0, stream>>>(lam, Scar);
    scan_apply_kernel<<<gScan, 256, 0, stream>>>(out, lam, Scar);
}
